// GraphNeuralNetwork_82961588290004
// MI455X (gfx1250) — matrix operand order repaired and checked by reading, not compiled, not run on hardware
//
#include <hip/hip_runtime.h>
#include <math.h>

#define NND 20000
#define NE  320000
#define FIN 64
#define HID 128
#define NH  4
#define NG  64
#define NOUT 32

typedef __attribute__((ext_vector_type(16))) __bf16          v16bf;
typedef __attribute__((ext_vector_type(16))) unsigned short  v16us;
typedef __attribute__((ext_vector_type(8)))  float           v8f;
typedef __attribute__((ext_vector_type(4)))  float           f32x4;

#define FLAG_RELU 1
#define FLAG_ACC  2

// ---------- helpers ----------
static __device__ __forceinline__ unsigned short f2bfu(float f) {
  return __builtin_bit_cast(unsigned short, (__bf16)f);
}
// order-preserving float -> uint key (for atomic max over signed floats)
static __device__ __forceinline__ unsigned fkey(float f) {
  unsigned u = __builtin_bit_cast(unsigned, f);
  return (u & 0x80000000u) ? ~u : (u | 0x80000000u);
}
static __device__ __forceinline__ float fkeyinv(unsigned k) {
  unsigned u = (k & 0x80000000u) ? (k & 0x7FFFFFFFu) : ~k;
  return __builtin_bit_cast(float, u);
}

// ---------- weight packing: f32 [K,Nc] row-major -> bf16 WMMA B-fragment order ----------
// fragment f = kc*(Nc/16)+nt holds B tile (k=kc*32..+31, n=nt*16..+15); per-lane 16 bf16:
// lane<16: k=kc*32+(j&7)+(j>>3)*16, n=nt*16+lane ; lane>=16: k=kc*32+8+(j&7)+(j>>3)*16, n=nt*16+lane-16
__global__ void pack_b_kernel(const float* __restrict__ W, unsigned short* __restrict__ P,
                              int K, int Nc) {
  int idx = blockIdx.x * 256 + threadIdx.x;
  if (idx >= K * Nc) return;
  int j    = idx & 15;
  int lane = (idx >> 4) & 31;
  int frag = idx >> 9;
  int nt   = frag % (Nc >> 4);
  int kc   = frag / (Nc >> 4);
  int k = kc * 32 + ((lane >> 4) << 3) + (j & 7) + ((j >> 3) << 4);
  int n = nt * 16 + (lane & 15);
  P[idx] = f2bfu(W[(size_t)k * Nc + n]);
}

// ---------- generic WMMA GEMM: out[M,Nc] = act(A[M,K]@W + bias [+ out]) ----------
template <int NT>
__global__ __launch_bounds__(32)
void wmma_gemm_kernel(const float* __restrict__ A, const unsigned short* __restrict__ Wp,
                      const float* __restrict__ bias, float* __restrict__ out,
                      int M, int K, int Nc, int flags) {
  const int lane = threadIdx.x & 31;
  const int half = lane >> 4;
  const int r    = lane & 15;
  const int mtile = blockIdx.x;
  const int row   = mtile * 16 + r;
  const int nbase = blockIdx.y * NT;   // in 16-col tiles
  const int nfr   = Nc >> 4;

  v8f zero = {0.f,0.f,0.f,0.f,0.f,0.f,0.f,0.f};
  v8f acc[NT];
#pragma unroll
  for (int t = 0; t < NT; t++) acc[t] = zero;

  for (int kc0 = 0; kc0 < K; kc0 += 32) {
    const float* ap = A + (size_t)row * K + kc0 + half * 8;
    if (kc0 + 32 < K) __builtin_prefetch(ap + 32, 0, 0);
    f32x4 a0 = *(const f32x4*)(ap);
    f32x4 a1 = *(const f32x4*)(ap + 4);
    f32x4 a2 = *(const f32x4*)(ap + 16);
    f32x4 a3 = *(const f32x4*)(ap + 20);
    v16bf a;
#pragma unroll
    for (int j = 0; j < 4; j++) {
      a[j]      = (__bf16)a0[j];
      a[4 + j]  = (__bf16)a1[j];
      a[8 + j]  = (__bf16)a2[j];
      a[12 + j] = (__bf16)a3[j];
    }
    const int kc = kc0 >> 5;
#pragma unroll
    for (int t = 0; t < NT; t++) {
      int frag = kc * nfr + nbase + t;
      v16us bu = *(const v16us*)(Wp + ((size_t)frag << 9) + lane * 16);
      v16bf b  = __builtin_bit_cast(v16bf, bu);
      acc[t] = __builtin_amdgcn_wmma_f32_16x16x32_bf16(
          false, a, false, b, (short)0, acc[t], false, false);
    }
  }

#pragma unroll
  for (int t = 0; t < NT; t++) {
    int col = (nbase + t) * 16 + r;
    float bv = bias ? bias[col] : 0.f;
#pragma unroll
    for (int j = 0; j < 8; j++) {
      int orow = mtile * 16 + j + half * 8;
      float* o = out + (size_t)orow * Nc + col;
      float v = acc[t][j] + bv;
      if (flags & FLAG_ACC)  v += *o;
      if (flags & FLAG_RELU) v = v > 0.f ? v : 0.f;
      *o = v;
    }
  }
}

// ---------- fused EdgeConv: per 16-edge tile: relu([xi|xj-xi]@W1+b1)@W2+b2 -> atomicMax ----------
__global__ __launch_bounds__(32)
void edgeconv_kernel(const float* __restrict__ H,
                     const int* __restrict__ src, const int* __restrict__ dst,
                     const unsigned short* __restrict__ W1p, const float* __restrict__ b1,
                     const unsigned short* __restrict__ W2p, const float* __restrict__ b2,
                     unsigned* __restrict__ outmax, int E) {
  __shared__ float lds[16 * 128];
  const int lane = threadIdx.x & 31;
  const int half = lane >> 4;
  const int r    = lane & 15;
  const int e    = blockIdx.x * 16 + r;
  const int s = src[e], d = dst[e];
  const float* hs = H + (size_t)s * HID;
  const float* hd = H + (size_t)d * HID;

  v8f zero = {0.f,0.f,0.f,0.f,0.f,0.f,0.f,0.f};
  v8f acc[8];
#pragma unroll
  for (int t = 0; t < 8; t++) acc[t] = zero;

  // Preload A fragments branch-free with aligned b128 loads:
  // chunks 0..3 hold x_i (= h[dst]); chunks 4..7 hold x_j - x_i.
  // Per-lane k-pattern for chunk kc: {kc*32+half*8 .. +7} and {kc*32+16+half*8 .. +7},
  // and chunks kc+4 use exactly the same per-lane indices of the other half of the
  // 256-wide input, so both halves share the same 8 x f32x4 row loads.
  v16bf afr[8];
#pragma unroll
  for (int kc = 0; kc < 4; kc++) {
    const float* pd = hd + kc * 32 + half * 8;
    const float* ps = hs + kc * 32 + half * 8;
    f32x4 d0 = *(const f32x4*)(pd);
    f32x4 d1 = *(const f32x4*)(pd + 4);
    f32x4 d2 = *(const f32x4*)(pd + 16);
    f32x4 d3 = *(const f32x4*)(pd + 20);
    f32x4 s0 = *(const f32x4*)(ps);
    f32x4 s1 = *(const f32x4*)(ps + 4);
    f32x4 s2 = *(const f32x4*)(ps + 16);
    f32x4 s3 = *(const f32x4*)(ps + 20);
#pragma unroll
    for (int j = 0; j < 4; j++) {
      afr[kc][j]          = (__bf16)d0[j];
      afr[kc][4 + j]      = (__bf16)d1[j];
      afr[kc][8 + j]      = (__bf16)d2[j];
      afr[kc][12 + j]     = (__bf16)d3[j];
      afr[kc + 4][j]      = (__bf16)(s0[j] - d0[j]);
      afr[kc + 4][4 + j]  = (__bf16)(s1[j] - d1[j]);
      afr[kc + 4][8 + j]  = (__bf16)(s2[j] - d2[j]);
      afr[kc + 4][12 + j] = (__bf16)(s3[j] - d3[j]);
    }
  }

  // GEMM1: 16x256x128, fully unrolled WMMA sweep
#pragma unroll
  for (int kc = 0; kc < 8; kc++) {
#pragma unroll
    for (int t = 0; t < 8; t++) {
      int frag = kc * 8 + t;
      v16us bu = *(const v16us*)(W1p + ((size_t)frag << 9) + lane * 16);
      v16bf b  = __builtin_bit_cast(v16bf, bu);
      acc[t] = __builtin_amdgcn_wmma_f32_16x16x32_bf16(
          false, afr[kc], false, b, (short)0, acc[t], false, false);
    }
  }
  // bias + relu -> LDS (row-major 16x128) for re-layout into A fragments
#pragma unroll
  for (int t = 0; t < 8; t++) {
    int col = t * 16 + r;
    float bv = b1[col];
#pragma unroll
    for (int j = 0; j < 8; j++) {
      float v = acc[t][j] + bv;
      lds[(j + half * 8) * 128 + col] = v > 0.f ? v : 0.f;
    }
  }
  __syncthreads();

  // GEMM2: 16x128x128, A fragments re-read from LDS via b128 loads
#pragma unroll
  for (int t = 0; t < 8; t++) acc[t] = zero;
#pragma unroll
  for (int kc = 0; kc < 4; kc++) {
    const float* lp = lds + r * 128 + kc * 32 + half * 8;
    f32x4 l0 = *(const f32x4*)(lp);
    f32x4 l1 = *(const f32x4*)(lp + 4);
    f32x4 l2 = *(const f32x4*)(lp + 16);
    f32x4 l3 = *(const f32x4*)(lp + 20);
    v16bf a;
#pragma unroll
    for (int j = 0; j < 4; j++) {
      a[j]      = (__bf16)l0[j];
      a[4 + j]  = (__bf16)l1[j];
      a[8 + j]  = (__bf16)l2[j];
      a[12 + j] = (__bf16)l3[j];
    }
#pragma unroll
    for (int t = 0; t < 8; t++) {
      int frag = kc * 8 + t;
      v16us bu = *(const v16us*)(W2p + ((size_t)frag << 9) + lane * 16);
      v16bf b  = __builtin_bit_cast(v16bf, bu);
      acc[t] = __builtin_amdgcn_wmma_f32_16x16x32_bf16(
          false, a, false, b, (short)0, acc[t], false, false);
    }
  }
  // bias, clamp >=0 (== relu(segmax) identity with 0-init), atomic max scatter
#pragma unroll
  for (int t = 0; t < 8; t++) {
    int col = t * 16 + r;
    float bv = b2[col];
#pragma unroll
    for (int j = 0; j < 8; j++) {
      int rr = j + half * 8;
      int dj = dst[blockIdx.x * 16 + rr];
      float v = acc[t][j] + bv;
      v = v > 0.f ? v : 0.f;   // non-negative: raw bits order as uint
      atomicMax(outmax + (size_t)dj * HID + col, __builtin_bit_cast(unsigned, v));
    }
  }
}

// ---------- elementwise / scatter kernels ----------
__global__ void fill_u32_kernel(unsigned* p, unsigned v, int n) {
  int i = blockIdx.x * 256 + threadIdx.x;
  if (i < n) p[i] = v;
}
__global__ void gat_alar_kernel(const float* __restrict__ HW, const float* __restrict__ as,
                                const float* __restrict__ ad, float* AL, float* AR, int Nn) {
  int i = blockIdx.x * 256 + threadIdx.x;
  if (i >= Nn * NH) return;
  int n = i >> 2, hh = i & 3;
  const float* hv = HW + (size_t)n * (NH * HID) + hh * HID;
  const float* a1 = as + hh * HID;
  const float* a2 = ad + hh * HID;
  float s1 = 0.f, s2 = 0.f;
  for (int k = 0; k < HID; k++) { float v = hv[k]; s1 += v * a1[k]; s2 += v * a2[k]; }
  AL[i] = s1; AR[i] = s2;
}
__global__ void gat_logits_kernel(const int* src, const int* dst, const float* AL, const float* AR,
                                  float* elog, unsigned* mx, int E, int Nn, int withSelf) {
  int Ecnt = withSelf ? E + Nn : E;
  int i = blockIdx.x * 256 + threadIdx.x;
  if (i >= Ecnt * NH) return;
  int e = i >> 2, hh = i & 3;
  int s, d;
  if (e < E) { s = src[e]; d = dst[e]; } else { s = d = e - E; }
  float l = AL[s * NH + hh] + AR[d * NH + hh];
  l = l > 0.f ? l : 0.2f * l;            // leaky relu
  elog[i] = l;
  atomicMax(mx + d * NH + hh, fkey(l));
}
__global__ void tf_logits_kernel(const int* src, const int* dst, const float* Q, const float* Kb,
                                 float* elog, unsigned* mx, int E) {
  int i = blockIdx.x * 256 + threadIdx.x;
  if (i >= E * NH) return;
  int e = i >> 2, hh = i & 3;
  int s = src[e], d = dst[e];
  const float* qv = Q  + (size_t)d * (NH * HID) + hh * HID;
  const float* kv = Kb + (size_t)s * (NH * HID) + hh * HID;
  float l = 0.f;
  for (int k = 0; k < HID; k++) l += qv[k] * kv[k];
  l *= 0.08838834764831845f;             // 1/sqrt(128)
  elog[i] = l;
  atomicMax(mx + d * NH + hh, fkey(l));
}
__global__ void edge_exp_kernel(const int* src, const int* dst, float* elog, const unsigned* mx,
                                float* den, int E, int Nn, int withSelf) {
  int Ecnt = withSelf ? E + Nn : E;
  int i = blockIdx.x * 256 + threadIdx.x;
  if (i >= Ecnt * NH) return;
  int e = i >> 2, hh = i & 3;
  int d = (e < E) ? dst[e] : e - E;
  float ex = expf(elog[i] - fkeyinv(mx[d * NH + hh]));
  elog[i] = ex;
  atomicAdd(den + d * NH + hh, ex);
}
__global__ void att_scatter_kernel(const int* src, const int* dst, const float* elog,
                                   const float* den, const float* V, float* out,
                                   int E, int Nn, int withSelf) {
  int Ecnt = withSelf ? E + Nn : E;
  int i = blockIdx.x * 256 + threadIdx.x;
  if (i >= Ecnt * HID) return;
  int e = i >> 7, dd = i & 127;
  int s, d;
  if (e < E) { s = src[e]; d = dst[e]; } else { s = d = e - E; }
  float val = 0.f;
#pragma unroll
  for (int hh = 0; hh < NH; hh++) {
    float alpha = elog[e * NH + hh] / den[d * NH + hh];
    val += alpha * V[(size_t)s * (NH * HID) + hh * HID + dd];
  }
  atomicAdd(out + (size_t)d * HID + dd, 0.25f * val);   // head mean folded in
}
__global__ void scatter_sum_kernel(const int* src, const int* dst, const float* H,
                                   float* out, int E) {
  int i = blockIdx.x * 256 + threadIdx.x;
  if (i >= E * HID) return;
  int e = i >> 7, dd = i & 127;
  atomicAdd(out + (size_t)dst[e] * HID + dd, H[(size_t)src[e] * HID + dd]);
}
__global__ void deg_kernel(const int* dst, float* deg, int E) {
  int i = blockIdx.x * 256 + threadIdx.x;
  if (i < E) atomicAdd(deg + dst[i], 1.f);
}
__global__ void rowdiv_kernel(float* h, const float* deg, int n) {
  int i = blockIdx.x * 256 + threadIdx.x;
  if (i < n) h[i] /= fmaxf(deg[i >> 7], 1.f);
}
__global__ void add_inplace_kernel(float* a, const float* b, int n) {
  int i = blockIdx.x * 256 + threadIdx.x;
  if (i < n) a[i] += b[i];
}
__global__ void bias_relu_kernel(float* h, const float* b, int n) {
  int i = blockIdx.x * 256 + threadIdx.x;
  if (i < n) { float v = h[i] + b[i & 127]; h[i] = v > 0.f ? v : 0.f; }
}
__global__ void bn_stats_kernel(const float* H, float* mu, float* rs, int Nn) {
  __shared__ float s1[256], s2[256];
  int c = blockIdx.x, t = threadIdx.x;
  float a = 0.f, b = 0.f;
  for (int n = t; n < Nn; n += 256) { float v = H[(size_t)n * HID + c]; a += v; b += v * v; }
  s1[t] = a; s2[t] = b; __syncthreads();
  for (int o = 128; o > 0; o >>= 1) {
    if (t < o) { s1[t] += s1[t + o]; s2[t] += s2[t + o]; }
    __syncthreads();
  }
  if (t == 0) {
    float m = s1[0] / Nn;
    float var = s2[0] / Nn - m * m;
    mu[c] = m; rs[c] = rsqrtf(var + 1e-5f);
  }
}
__global__ void bn_apply_kernel(float* h, const float* g, const float* b,
                                const float* mu, const float* rs, int n) {
  int i = blockIdx.x * 256 + threadIdx.x;
  if (i >= n) return;
  int c = i & 127;
  h[i] = g[c] * (h[i] - mu[c]) * rs[c] + b[c];
}
__global__ void pool_cnt_kernel(const int* batch, float* cnt, int Nn) {
  int i = blockIdx.x * 256 + threadIdx.x;
  if (i < Nn) atomicAdd(cnt + batch[i], 1.f);
}
__global__ void pool_sum_kernel(const int* batch, const float* h, float* pool, int Nn) {
  int i = blockIdx.x * 256 + threadIdx.x;
  if (i >= Nn * HID) return;
  atomicAdd(pool + (size_t)batch[i >> 7] * HID + (i & 127), h[i]);
}

// ---------- host ----------
extern "C" void kernel_launch(void* const* d_in, const int* in_sizes, int n_in,
                              void* d_out, int out_size, void* d_ws, size_t ws_size,
                              hipStream_t stream) {
  (void)in_sizes; (void)n_in; (void)out_size; (void)ws_size;
  const int Esl = NE + NND;

  const float* x     = (const float*)d_in[0];
  const int*   ei    = (const int*)d_in[1];
  const int*   src   = ei;
  const int*   dst   = ei + NE;
  const int*   batch = (const int*)d_in[2];
  const float *g1_w=(const float*)d_in[3],  *g1_as=(const float*)d_in[4],
              *g1_ad=(const float*)d_in[5], *g1_b=(const float*)d_in[6],
              *g2_w=(const float*)d_in[7],  *g2_as=(const float*)d_in[8],
              *g2_ad=(const float*)d_in[9], *g2_b=(const float*)d_in[10],
              *bn1_g=(const float*)d_in[11],*bn1_b=(const float*)d_in[12],
              *bn2_g=(const float*)d_in[13],*bn2_b=(const float*)d_in[14],
              *gin_w1=(const float*)d_in[15],*gin_b1=(const float*)d_in[16],
              *gin_w2=(const float*)d_in[17],*gin_b2=(const float*)d_in[18],
              *s1_wl=(const float*)d_in[19],*s1_bl=(const float*)d_in[20],*s1_wr=(const float*)d_in[21],
              *s2_wl=(const float*)d_in[22],*s2_bl=(const float*)d_in[23],*s2_wr=(const float*)d_in[24],
              *t_wq=(const float*)d_in[25],*t_bq=(const float*)d_in[26],
              *t_wk=(const float*)d_in[27],*t_bk=(const float*)d_in[28],
              *t_wv=(const float*)d_in[29],*t_bv=(const float*)d_in[30],
              *t_ws=(const float*)d_in[31],*t_bs=(const float*)d_in[32],
              *e_w1=(const float*)d_in[33],*e_b1=(const float*)d_in[34],
              *e_w2=(const float*)d_in[35],*e_b2=(const float*)d_in[36],
              *fc1_w=(const float*)d_in[37],*fc1_b=(const float*)d_in[38],
              *fc2_w=(const float*)d_in[39],*fc2_b=(const float*)d_in[40];

  char* base = (char*)d_ws;
  size_t off = 0;
  auto alloc = [&](size_t bytes) -> void* {
    void* p = base + off;
    off = (off + bytes + 255) & ~(size_t)255;
    return p;
  };
  // packed bf16 weights
  unsigned short* Pg1  = (unsigned short*)alloc(64  * 512 * 2);
  unsigned short* Pg2  = (unsigned short*)alloc(128 * 512 * 2);
  unsigned short* Pgin1= (unsigned short*)alloc(128 * 128 * 2);
  unsigned short* Pgin2= (unsigned short*)alloc(128 * 128 * 2);
  unsigned short* Ps1l = (unsigned short*)alloc(128 * 128 * 2);
  unsigned short* Ps1r = (unsigned short*)alloc(128 * 128 * 2);
  unsigned short* Ps2l = (unsigned short*)alloc(128 * 128 * 2);
  unsigned short* Ps2r = (unsigned short*)alloc(128 * 128 * 2);
  unsigned short* Pq   = (unsigned short*)alloc(128 * 512 * 2);
  unsigned short* Pk   = (unsigned short*)alloc(128 * 512 * 2);
  unsigned short* Pv   = (unsigned short*)alloc(128 * 512 * 2);
  unsigned short* Pts  = (unsigned short*)alloc(128 * 128 * 2);
  unsigned short* Pe1  = (unsigned short*)alloc(256 * 128 * 2);
  unsigned short* Pe2  = (unsigned short*)alloc(128 * 128 * 2);
  unsigned short* Pf1  = (unsigned short*)alloc(128 * 128 * 2);
  unsigned short* Pf2  = (unsigned short*)alloc(128 * 32  * 2);
  // activations / scratch
  float* HW  = (float*)alloc((size_t)NND * 512 * 4);   // GAT features / Q
  float* KB  = (float*)alloc((size_t)NND * 512 * 4);
  float* VB  = (float*)alloc((size_t)NND * 512 * 4);
  float* X1  = (float*)alloc((size_t)NND * HID * 4);
  float* X2  = (float*)alloc((size_t)NND * HID * 4);
  float* X3  = (float*)alloc((size_t)NND * HID * 4);
  float* AL  = (float*)alloc((size_t)NND * NH * 4);
  float* AR  = (float*)alloc((size_t)NND * NH * 4);
  float* ELOG= (float*)alloc((size_t)Esl * NH * 4);
  unsigned* MX = (unsigned*)alloc((size_t)NND * NH * 4);
  float* DEN = (float*)alloc((size_t)NND * NH * 4);
  float* DEG = (float*)alloc((size_t)NND * 4);
  float* MU  = (float*)alloc(HID * 4);
  float* RS  = (float*)alloc(HID * 4);
  float* POOL= (float*)alloc((size_t)NG * HID * 4);
  float* CNT = (float*)alloc((size_t)NG * 4);
  float* PH  = (float*)alloc((size_t)NG * HID * 4);

  auto cdiv = [](int a, int b) { return (a + b - 1) / b; };
  auto pack = [&](const float* W, unsigned short* P, int K, int Nc) {
    pack_b_kernel<<<cdiv(K * Nc, 256), 256, 0, stream>>>(W, P, K, Nc);
  };
  auto gemm4 = [&](const float* A, const unsigned short* Wp, const float* bias, float* out,
                   int M, int K, int Nc, int flags) {
    dim3 g(M / 16, Nc / 64);
    wmma_gemm_kernel<4><<<g, 32, 0, stream>>>(A, Wp, bias, out, M, K, Nc, flags);
  };
  auto fill = [&](void* p, unsigned v, int n) {
    fill_u32_kernel<<<cdiv(n, 256), 256, 0, stream>>>((unsigned*)p, v, n);
  };

  // pack all weights
  pack(g1_w, Pg1, 64, 512);   pack(g2_w, Pg2, 128, 512);
  pack(gin_w1, Pgin1, 128, 128); pack(gin_w2, Pgin2, 128, 128);
  pack(s1_wl, Ps1l, 128, 128); pack(s1_wr, Ps1r, 128, 128);
  pack(s2_wl, Ps2l, 128, 128); pack(s2_wr, Ps2r, 128, 128);
  pack(t_wq, Pq, 128, 512); pack(t_wk, Pk, 128, 512); pack(t_wv, Pv, 128, 512);
  pack(t_ws, Pts, 128, 128);
  pack(e_w1, Pe1, 256, 128); pack(e_w2, Pe2, 128, 128);
  pack(fc1_w, Pf1, 128, 128); pack(fc2_w, Pf2, 128, 32);

  // ---- GAT layer 1 ----
  gemm4(x, Pg1, nullptr, HW, NND, 64, 512, 0);
  gat_alar_kernel<<<cdiv(NND * NH, 256), 256, 0, stream>>>(HW, g1_as, g1_ad, AL, AR, NND);
  fill(MX, 0u, NND * NH); fill(DEN, 0u, NND * NH);
  gat_logits_kernel<<<cdiv(Esl * NH, 256), 256, 0, stream>>>(src, dst, AL, AR, ELOG, MX, NE, NND, 1);
  edge_exp_kernel<<<cdiv(Esl * NH, 256), 256, 0, stream>>>(src, dst, ELOG, MX, DEN, NE, NND, 1);
  fill(X1, 0u, NND * HID);
  att_scatter_kernel<<<cdiv(Esl * HID, 256), 256, 0, stream>>>(src, dst, ELOG, DEN, HW, X1, NE, NND, 1);
  bias_relu_kernel<<<cdiv(NND * HID, 256), 256, 0, stream>>>(X1, g1_b, NND * HID);
  bn_stats_kernel<<<HID, 256, 0, stream>>>(X1, MU, RS, NND);
  bn_apply_kernel<<<cdiv(NND * HID, 256), 256, 0, stream>>>(X1, bn1_g, bn1_b, MU, RS, NND * HID);

  // ---- GAT layer 2 ----
  gemm4(X1, Pg2, nullptr, HW, NND, 128, 512, 0);
  gat_alar_kernel<<<cdiv(NND * NH, 256), 256, 0, stream>>>(HW, g2_as, g2_ad, AL, AR, NND);
  fill(MX, 0u, NND * NH); fill(DEN, 0u, NND * NH);
  gat_logits_kernel<<<cdiv(Esl * NH, 256), 256, 0, stream>>>(src, dst, AL, AR, ELOG, MX, NE, NND, 1);
  edge_exp_kernel<<<cdiv(Esl * NH, 256), 256, 0, stream>>>(src, dst, ELOG, MX, DEN, NE, NND, 1);
  fill(X2, 0u, NND * HID);
  att_scatter_kernel<<<cdiv(Esl * HID, 256), 256, 0, stream>>>(src, dst, ELOG, DEN, HW, X2, NE, NND, 1);
  bias_relu_kernel<<<cdiv(NND * HID, 256), 256, 0, stream>>>(X2, g2_b, NND * HID);
  bn_stats_kernel<<<HID, 256, 0, stream>>>(X2, MU, RS, NND);
  bn_apply_kernel<<<cdiv(NND * HID, 256), 256, 0, stream>>>(X2, bn2_g, bn2_b, MU, RS, NND * HID);

  // ---- GIN ----
  fill(X3, 0u, NND * HID);
  scatter_sum_kernel<<<cdiv(NE * HID, 256), 256, 0, stream>>>(src, dst, X2, X3, NE);
  add_inplace_kernel<<<cdiv(NND * HID, 256), 256, 0, stream>>>(X3, X2, NND * HID);
  gemm4(X3, Pgin1, gin_b1, X1, NND, 128, 128, FLAG_RELU);
  gemm4(X1, Pgin2, gin_b2, X3, NND, 128, 128, FLAG_RELU);   // relu(g)

  // ---- degrees ----
  fill(DEG, 0u, NND);
  deg_kernel<<<cdiv(NE, 256), 256, 0, stream>>>(dst, DEG, NE);

  // ---- SAGE layer 1 ----
  fill(X1, 0u, NND * HID);
  scatter_sum_kernel<<<cdiv(NE * HID, 256), 256, 0, stream>>>(src, dst, X3, X1, NE);
  rowdiv_kernel<<<cdiv(NND * HID, 256), 256, 0, stream>>>(X1, DEG, NND * HID);
  gemm4(X1, Ps1l, s1_bl, X2, NND, 128, 128, 0);
  gemm4(X3, Ps1r, nullptr, X2, NND, 128, 128, FLAG_ACC | FLAG_RELU);

  // ---- SAGE layer 2 ----
  fill(X1, 0u, NND * HID);
  scatter_sum_kernel<<<cdiv(NE * HID, 256), 256, 0, stream>>>(src, dst, X2, X1, NE);
  rowdiv_kernel<<<cdiv(NND * HID, 256), 256, 0, stream>>>(X1, DEG, NND * HID);
  gemm4(X1, Ps2l, s2_bl, X3, NND, 128, 128, 0);
  gemm4(X2, Ps2r, nullptr, X3, NND, 128, 128, FLAG_ACC | FLAG_RELU);

  // ---- TransformerConv ----
  gemm4(X3, Pq, t_bq, HW, NND, 128, 512, 0);
  gemm4(X3, Pk, t_bk, KB, NND, 128, 512, 0);
  gemm4(X3, Pv, t_bv, VB, NND, 128, 512, 0);
  fill(MX, 0u, NND * NH); fill(DEN, 0u, NND * NH);
  tf_logits_kernel<<<cdiv(NE * NH, 256), 256, 0, stream>>>(src, dst, HW, KB, ELOG, MX, NE);
  edge_exp_kernel<<<cdiv(NE * NH, 256), 256, 0, stream>>>(src, dst, ELOG, MX, DEN, NE, NND, 0);
  fill(X1, 0u, NND * HID);
  att_scatter_kernel<<<cdiv(NE * HID, 256), 256, 0, stream>>>(src, dst, ELOG, DEN, VB, X1, NE, NND, 0);
  gemm4(X3, Pts, t_bs, X1, NND, 128, 128, FLAG_ACC | FLAG_RELU);

  // ---- EdgeConv (fused WMMA MLP + atomic max) ----
  fill(X2, 0u, NND * HID);   // 0.0f == uint 0 : identity for non-negative max
  edgeconv_kernel<<<NE / 16, 32, 0, stream>>>(X1, src, dst, Pe1, e_b1, Pe2, e_b2,
                                              (unsigned*)X2, NE);

  // ---- global mean pool + heads ----
  fill(POOL, 0u, NG * HID); fill(CNT, 0u, NG);
  pool_cnt_kernel<<<cdiv(NND, 256), 256, 0, stream>>>(batch, CNT, NND);
  pool_sum_kernel<<<cdiv(NND * HID, 256), 256, 0, stream>>>(batch, X2, POOL, NND);
  rowdiv_kernel<<<cdiv(NG * HID, 256), 256, 0, stream>>>(POOL, CNT, NG * HID);
  gemm4(POOL, Pf1, fc1_b, PH, NG, 128, 128, FLAG_RELU);
  {
    dim3 g(NG / 16, 1);
    wmma_gemm_kernel<2><<<g, 32, 0, stream>>>(PH, Pf2, fc2_b, (float*)d_out,
                                              NG, 128, 32, 0);
  }
}
